// MultiHeadedAttentionSANM_61546881352159
// MI455X (gfx1250) — compile-verified, matrix operand order checked by reading, not run on hardware
//
#include <hip/hip_runtime.h>
#include <math.h>

typedef __attribute__((ext_vector_type(16))) __bf16 v16bf;
typedef __attribute__((ext_vector_type(8)))  float  v8f;

#define B_  8
#define T_  1024
#define F_  1024
#define HQ  16
#define DK  64
#define F3  3072
#define KW  11
#define LPAD 5

static __device__ __forceinline__ v8f wmma_bf16(v16bf a, v16bf b, v8f c) {
  return __builtin_amdgcn_wmma_f32_16x16x32_bf16(false, a, false, b, (short)0, c, false, false);
}

// ---------------- f32 -> bf16 conversion ----------------
__global__ void k_f32_to_bf16(const float* __restrict__ in, __bf16* __restrict__ out, int n) {
  int i = blockIdx.x * blockDim.x + threadIdx.x;
  if (i < n) out[i] = (__bf16)in[i];
}

// ---------------- QKV GEMM: C[8192,3072] = A[8192,1024] x W[3072,1024]^T ----------------
// each wave computes a 64x64 tile; K loop in steps of 32; bf16 in, f32 acc, bf16 out (+bias)
__global__ __launch_bounds__(256) void k_gemm_qkv(const __bf16* __restrict__ A,
                                                  const __bf16* __restrict__ W,
                                                  const float*  __restrict__ bias,
                                                  __bf16* __restrict__ C) {
  const int lane = threadIdx.x & 31;
  const int wid  = (int)((blockIdx.x * blockDim.x + threadIdx.x) >> 5);
  const int tilesN = F3 / 64;
  const int tm = wid / tilesN, tn = wid % tilesN;
  const int m0 = tm * 64, n0 = tn * 64;
  const int lm = lane & 15, lh = lane >> 4;

  v8f acc[4][4];
  for (int i = 0; i < 4; i++) for (int j = 0; j < 4; j++) acc[i][j] = {};

  for (int k0 = 0; k0 < F_; k0 += 32) {
    v16bf a[4], b[4];
    for (int i = 0; i < 4; i++)
      a[i] = *(const v16bf*)(A + (size_t)(m0 + i * 16 + lm) * F_ + k0 + lh * 16);
    for (int j = 0; j < 4; j++)
      b[j] = *(const v16bf*)(W + (size_t)(n0 + j * 16 + lm) * F_ + k0 + lh * 16);
    for (int i = 0; i < 4; i++)
      for (int j = 0; j < 4; j++)
        acc[i][j] = wmma_bf16(a[i], b[j], acc[i][j]);
  }
  for (int j = 0; j < 4; j++) {
    const int col = n0 + j * 16 + lm;
    const float bj = bias[col];
    for (int i = 0; i < 4; i++)
      for (int r = 0; r < 8; r++) {
        const int row = m0 + i * 16 + r + 8 * lh;
        C[(size_t)row * F3 + col] = (__bf16)(acc[i][j][r] + bj);
      }
  }
}

// ---------------- Output GEMM: out[8192,1024] = ctx x Wout^T + bias + fsmn(in d_out) ----------------
__global__ __launch_bounds__(256) void k_gemm_out(const __bf16* __restrict__ A,
                                                  const __bf16* __restrict__ W,
                                                  const float*  __restrict__ bias,
                                                  float* __restrict__ out) {
  const int lane = threadIdx.x & 31;
  const int wid  = (int)((blockIdx.x * blockDim.x + threadIdx.x) >> 5);
  const int tilesN = F_ / 64;
  const int tm = wid / tilesN, tn = wid % tilesN;
  const int m0 = tm * 64, n0 = tn * 64;
  const int lm = lane & 15, lh = lane >> 4;

  v8f acc[4][4];
  for (int i = 0; i < 4; i++) for (int j = 0; j < 4; j++) acc[i][j] = {};

  for (int k0 = 0; k0 < F_; k0 += 32) {
    v16bf a[4], b[4];
    for (int i = 0; i < 4; i++)
      a[i] = *(const v16bf*)(A + (size_t)(m0 + i * 16 + lm) * F_ + k0 + lh * 16);
    for (int j = 0; j < 4; j++)
      b[j] = *(const v16bf*)(W + (size_t)(n0 + j * 16 + lm) * F_ + k0 + lh * 16);
    for (int i = 0; i < 4; i++)
      for (int j = 0; j < 4; j++)
        acc[i][j] = wmma_bf16(a[i], b[j], acc[i][j]);
  }
  for (int j = 0; j < 4; j++) {
    const int col = n0 + j * 16 + lm;
    const float bj = bias[col];
    for (int i = 0; i < 4; i++)
      for (int r = 0; r < 8; r++) {
        const size_t idx = (size_t)(m0 + i * 16 + r + 8 * lh) * F_ + col;
        out[idx] = acc[i][j][r] + bj + out[idx];   // fsmn already in d_out
      }
  }
}

// ---------------- V transpose: vt[b,h,d,t] = qkv[b,t, 2048 + h*64+d] ----------------
__global__ void k_vt(const __bf16* __restrict__ qkv, __bf16* __restrict__ vt) {
  int i = blockIdx.x * blockDim.x + threadIdx.x;
  if (i >= B_ * T_ * F_) return;
  int d = i & (F_ - 1); int bt = i >> 10; int t = bt & (T_ - 1); int b = bt >> 10;
  int h = d >> 6; int dd = d & 63;
  vt[((size_t)(b * HQ + h) * DK + dd) * T_ + t] = qkv[(size_t)bt * F3 + 2 * F_ + d];
}

// ---------------- FSMN depthwise conv (f32) -> writes d_out ----------------
__global__ void k_fsmn(const __bf16* __restrict__ qkv, const int* __restrict__ mask,
                       const float* __restrict__ kern, float* __restrict__ out) {
  int i = blockIdx.x * blockDim.x + threadIdx.x;
  if (i >= B_ * T_ * F_) return;
  int d = i & (F_ - 1); int bt = i >> 10; int t = bt & (T_ - 1); int b = bt >> 10;
  if (mask[b * T_ + t] == 0) { out[i] = 0.f; return; }
  float acc = (float)qkv[(size_t)bt * F3 + 2 * F_ + d];   // + vm (center, mask==1)
  for (int j = 0; j < KW; j++) {
    int tt = t - LPAD + j;
    if (tt < 0 || tt >= T_) continue;
    if (mask[b * T_ + tt] == 0) continue;
    float v = (float)qkv[(size_t)(b * T_ + tt) * F3 + 2 * F_ + d];
    acc += kern[d * KW + j] * v;
  }
  out[i] = acc;
}

// ---------------- Flash attention: 1 wave = 16 queries, key tiles of 32 ----------------
__global__ __launch_bounds__(128) void k_attn(const __bf16* __restrict__ qkv,
                                              const __bf16* __restrict__ vt,
                                              const int*    __restrict__ mask,
                                              __bf16* __restrict__ ctx) {
  __shared__ __align__(32) __bf16 pt[4][16][32];   // per-wave P tile (C-layout -> A-layout)
  const int lane = threadIdx.x & 31;
  const int w    = threadIdx.x >> 5;
  const int bh = blockIdx.x; const int b = bh >> 4; const int h = bh & 15;
  const int q0 = blockIdx.y * 64 + w * 16;
  const int lm = lane & 15, lh = lane >> 4;
  const int* mb = mask + b * T_;
  const float scale = 0.125f;           // 1/sqrt(64)
  const float NEG = -INFINITY;

  // Q fragments (A-layout, K = head dim, two 32-wide steps)
  const __bf16* qrow = qkv + (size_t)(b * T_ + q0 + lm) * F3 + h * DK;
  v16bf aq0 = *(const v16bf*)(qrow + lh * 16);
  v16bf aq1 = *(const v16bf*)(qrow + 32 + lh * 16);

  v8f acc[4]; for (int dt = 0; dt < 4; dt++) acc[dt] = {};
  float mi[8], li[8];
  for (int r = 0; r < 8; r++) { mi[r] = NEG; li[r] = 0.f; }

  const __bf16* kbase = qkv + (size_t)(b * T_) * F3 + F_ + h * DK;
  const __bf16* vb    = vt  + (size_t)(b * HQ + h) * DK * T_;

  for (int kt = 0; kt < T_; kt += 32) {
    // K fragments (B-layout: n = key, K = head dim)
    v16bf bk00 = *(const v16bf*)(kbase + (size_t)(kt + lm) * F3 + lh * 16);
    v16bf bk01 = *(const v16bf*)(kbase + (size_t)(kt + lm) * F3 + 32 + lh * 16);
    v16bf bk10 = *(const v16bf*)(kbase + (size_t)(kt + 16 + lm) * F3 + lh * 16);
    v16bf bk11 = *(const v16bf*)(kbase + (size_t)(kt + 16 + lm) * F3 + 32 + lh * 16);
    v8f s0 = {}, s1 = {};
    s0 = wmma_bf16(aq0, bk00, s0); s0 = wmma_bf16(aq1, bk01, s0);
    s1 = wmma_bf16(aq0, bk10, s1); s1 = wmma_bf16(aq1, bk11, s1);

    const int msk0 = mb[kt + lm], msk1 = mb[kt + 16 + lm];
    float fac[8], p0v[8], p1v[8];
    for (int r = 0; r < 8; r++) {
      float x0 = msk0 ? s0[r] * scale : NEG;
      float x1 = msk1 ? s1[r] * scale : NEG;
      float mx = fmaxf(x0, x1);
      for (int off = 8; off >= 1; off >>= 1) mx = fmaxf(mx, __shfl_xor(mx, off, 16));
      float mnew = fmaxf(mi[r], mx);
      float f, p0, p1;
      if (mnew == NEG) { f = 0.f; p0 = 0.f; p1 = 0.f; }
      else {
        f  = (mi[r] == NEG) ? 0.f : __expf(mi[r] - mnew);
        p0 = (x0 == NEG) ? 0.f : __expf(x0 - mnew);
        p1 = (x1 == NEG) ? 0.f : __expf(x1 - mnew);
      }
      float sum = p0 + p1;
      for (int off = 8; off >= 1; off >>= 1) sum += __shfl_xor(sum, off, 16);
      li[r] = li[r] * f + sum;
      mi[r] = mnew;
      fac[r] = f; p0v[r] = p0; p1v[r] = p1;
    }
    for (int dt = 0; dt < 4; dt++)
      for (int r = 0; r < 8; r++) acc[dt][r] *= fac[r];

    // C-layout -> A-layout transpose of P through per-wave LDS tile (wave-synchronous,
    // DS ops are in-order within a wave; explicit s_wait_dscnt before readback)
    for (int r = 0; r < 8; r++) {
      pt[w][r + 8 * lh][lm]      = (__bf16)p0v[r];
      pt[w][r + 8 * lh][16 + lm] = (__bf16)p1v[r];
    }
    asm volatile("s_wait_dscnt 0x0" ::: "memory");
    v16bf pf = *(const v16bf*)&pt[w][lm][lh * 16];

    // P x V accumulate (V^T is contiguous along keys)
    for (int dt = 0; dt < 4; dt++) {
      v16bf bv = *(const v16bf*)(vb + (size_t)(dt * 16 + lm) * T_ + kt + lh * 16);
      acc[dt] = wmma_bf16(pf, bv, acc[dt]);
    }
  }

  for (int dt = 0; dt < 4; dt++)
    for (int r = 0; r < 8; r++) {
      const int row = b * T_ + q0 + r + 8 * lh;
      const int col = h * DK + dt * 16 + lm;
      float o = (li[r] > 0.f) ? acc[dt][r] / li[r] : 0.f;
      ctx[(size_t)row * F_ + col] = (__bf16)o;
    }
}

extern "C" void kernel_launch(void* const* d_in, const int* in_sizes, int n_in,
                              void* d_out, int out_size, void* d_ws, size_t ws_size,
                              hipStream_t stream) {
  (void)in_sizes; (void)n_in; (void)out_size; (void)ws_size;
  const float* hidden = (const float*)d_in[0];
  const int*   mask   = (const int*)  d_in[1];
  const float* Wqkv   = (const float*)d_in[2];
  const float* bqkv   = (const float*)d_in[3];
  const float* Wout   = (const float*)d_in[4];
  const float* bout   = (const float*)d_in[5];
  const float* fker   = (const float*)d_in[6];
  float* out = (float*)d_out;

  char* ws = (char*)d_ws;
  __bf16* Xbf   = (__bf16*)ws; ws += (size_t)B_ * T_ * F_ * 2;
  __bf16* Wqkvb = (__bf16*)ws; ws += (size_t)F3 * F_ * 2;
  __bf16* Woutb = (__bf16*)ws; ws += (size_t)F_ * F_ * 2;
  __bf16* qkvb  = (__bf16*)ws; ws += (size_t)B_ * T_ * F3 * 2;
  __bf16* vtb   = (__bf16*)ws; ws += (size_t)B_ * T_ * F_ * 2;
  __bf16* ctxb  = (__bf16*)ws; ws += (size_t)B_ * T_ * F_ * 2;

  const int nHid = B_ * T_ * F_;        // 8388608
  const int nWq  = F3 * F_;             // 3145728
  const int nWo  = F_ * F_;             // 1048576

  k_f32_to_bf16<<<(nHid + 255) / 256, 256, 0, stream>>>(hidden, Xbf, nHid);
  k_f32_to_bf16<<<(nWq  + 255) / 256, 256, 0, stream>>>(Wqkv, Wqkvb, nWq);
  k_f32_to_bf16<<<(nWo  + 255) / 256, 256, 0, stream>>>(Wout, Woutb, nWo);

  // QKV projection: 128 x 48 = 6144 wave-tiles, 8 waves / block
  k_gemm_qkv<<<768, 256, 0, stream>>>(Xbf, Wqkvb, bqkv, qkvb);

  k_vt  <<<(nHid + 255) / 256, 256, 0, stream>>>(qkvb, vtb);
  k_fsmn<<<(nHid + 255) / 256, 256, 0, stream>>>(qkvb, mask, fker, out);

  dim3 ag(B_ * HQ, T_ / 64);            // 128 x 16 blocks, 4 waves each
  k_attn<<<ag, 128, 0, stream>>>(qkvb, vtb, mask, ctxb);

  // Output projection: 128 x 16 = 2048 wave-tiles
  k_gemm_out<<<256, 256, 0, stream>>>(ctxb, Woutb, bout, out);
}
